// RnnCoreModel_10642928959712
// MI455X (gfx1250) — matrix operand order repaired and checked by reading, not compiled, not run on hardware
//
#include <hip/hip_runtime.h>
#include <hip/hip_bf16.h>

// ---------------- problem constants ----------------
#define B_     32
#define S_     512
#define IN_    512
#define AD_    64
#define PROJ_  960          // INNER - ACTION_DIM
#define INNER_ 1024
#define H_     1024
#define G4_    4096         // 4*H
#define MROWS_ (B_ * S_)    // 16384
#define RWG_   32           // workgroups in the persistent recurrent kernel

typedef __attribute__((ext_vector_type(16))) __bf16 v16bf;
typedef __attribute__((ext_vector_type(8)))  float  v8f;

union TileBf { uint4 q[2]; v16bf v; };   // 32 bytes

__device__ __forceinline__ unsigned short f32_bf16(float f) {
  unsigned u = __float_as_uint(f);
  u += 0x7FFFu + ((u >> 16) & 1u);       // round-to-nearest-even
  return (unsigned short)(u >> 16);
}
__device__ __forceinline__ float bf16_f32(unsigned short h) {
  return __uint_as_float(((unsigned)h) << 16);
}

// A-tile (16x32, bf16, M rows x K): per ISA 7.12.2 lane l<16 holds row l,
// K {k0..k0+7, k0+16..k0+23}; lanes 16-31 hold the +8 shifted K halves.
__device__ __forceinline__ v16bf load_a16x32(const unsigned short* base, long ld, int lane) {
  const unsigned short* p = base + (long)(lane & 15) * ld + ((lane >> 4) ? 8 : 0);
  TileBf t;
  t.q[0] = *(const uint4*)(p);
  t.q[1] = *(const uint4*)(p + 16);
  return t.v;
}
// B-tile (32x16 = K x N) sourced from a row-major (N,K) weight matrix:
// lane l holds column n = n0 + l%16, K {k0+(l/16)*8 .. +8, k0+16+(l/16)*8 .. +8}.
__device__ __forceinline__ v16bf load_b32x16(const unsigned short* base, long ld, int lane) {
  const unsigned short* p = base + (long)(lane & 15) * ld + ((lane >> 4) ? 8 : 0);
  TileBf t;
  t.q[0] = *(const uint4*)(p);
  t.q[1] = *(const uint4*)(p + 16);
  return t.v;
}

// ---------------- prep kernels ----------------
__global__ void k_cvt_bf16(const float* __restrict__ s, unsigned short* __restrict__ d, long n) {
  long i = (long)blockIdx.x * blockDim.x + threadIdx.x;
  long st = (long)gridDim.x * blockDim.x;
  for (; i < n; i += st) d[i] = f32_bf16(s[i]);
}

__global__ void k_embed(const int* __restrict__ act, const float* __restrict__ emb,
                        unsigned short* __restrict__ x0) {
  long n = (long)MROWS_ * AD_;
  long i = (long)blockIdx.x * blockDim.x + threadIdx.x;
  long st = (long)gridDim.x * blockDim.x;
  for (; i < n; i += st) {
    long row = i >> 6; int c = (int)(i & 63);
    x0[row * INNER_ + PROJ_ + c] = f32_bf16(emb[(long)act[row] * AD_ + c]);
  }
}

__global__ void k_zero_bar(unsigned int* bar) { if (threadIdx.x == 0) bar[0] = 0u; }

// ---------------- generic bf16 WMMA GEMM: C(M,N) = act(A(M,K) @ Bw(N,K)^T + bias) ----------------
// block = 128 (4 waves); each wave computes a 32(M) x 64(N) tile:
// per k-step 2 A-tiles + 4 B-tiles feed 8 WMMAs (A reused across N, B across M).
// grid = (N/64, M/128).
__global__ void k_gemm_bf16(const unsigned short* __restrict__ A, int lda,
                            const unsigned short* __restrict__ Bw, int ldb,
                            const float* __restrict__ bias0, const float* __restrict__ bias1,
                            unsigned short* __restrict__ C, int ldc,
                            int K, int relu) {
  const int lane = threadIdx.x & 31;
  const int wave = threadIdx.x >> 5;
  const long m0 = (long)blockIdx.y * 128 + (long)wave * 32;
  const long n0 = (long)blockIdx.x * 64;
  v8f acc[2][4] = {};
  for (int k = 0; k < K; k += 32) {
    v16bf a0 = load_a16x32(A + m0 * lda + k, lda, lane);
    v16bf a1 = load_a16x32(A + (m0 + 16) * lda + k, lda, lane);
#pragma unroll
    for (int nt = 0; nt < 4; ++nt) {
      v16bf b = load_b32x16(Bw + (n0 + nt * 16) * (long)ldb + k, ldb, lane);
      acc[0][nt] = __builtin_amdgcn_wmma_f32_16x16x32_bf16(false, a0, false, b,
                                                           (short)0, acc[0][nt], false, false);
      acc[1][nt] = __builtin_amdgcn_wmma_f32_16x16x32_bf16(false, a1, false, b,
                                                           (short)0, acc[1][nt], false, false);
    }
  }
  // C layout (ISA 7.12.2): lane l, vgpr r -> (m = r + 8*(l/16), n = l%16)
  const int mh = (lane >> 4) * 8;
#pragma unroll
  for (int mt = 0; mt < 2; ++mt) {
#pragma unroll
    for (int nt = 0; nt < 4; ++nt) {
      long n = n0 + nt * 16 + (lane & 15);
      float bs = (bias0 ? bias0[n] : 0.f) + (bias1 ? bias1[n] : 0.f);
#pragma unroll
      for (int r = 0; r < 8; ++r) {
        float v = acc[mt][nt][r] + bs;
        if (relu) v = fmaxf(v, 0.f);
        C[(m0 + mt * 16 + r + mh) * (long)ldc + n] = f32_bf16(v);
      }
    }
  }
}

// ---------------- persistent LSTM recurrence ----------------
// grid = 32 WGs x 256 threads. WG wg owns h-columns j0=wg*32 .. +32 and the four
// gate rows {g*1024 + j} for those columns. Its 128x1024 bf16 W_hh slice (256KB)
// is DMA'd into LDS ONCE via GLOBAL_LOAD_ASYNC_TO_LDS (ASYNCcnt) and reused
// across all 512 steps (CDNA5 320KB LDS). c-state lives in LDS for the whole
// sequence. h is exchanged via L2 + a grid generation barrier.
__global__ void k_lstm_recur(const unsigned short* __restrict__ xg,   // (16384,4096) bf16, bias folded
                             const unsigned short* __restrict__ whh,  // (4096,1024) bf16, this layer
                             const unsigned short* __restrict__ hsrc, // bf16 h storage (read)
                             unsigned short* __restrict__ hdst,       // bf16 h storage (write)
                             int hmode,                               // 0: row=b*S+t  1: row=(t&1)*32+b
                             float* __restrict__ out_h, float* __restrict__ out_c,
                             float* __restrict__ out_h2,              // rnn_out alias (layer 1) or null
                             unsigned int* __restrict__ bar) {
  extern __shared__ char smem[];
  unsigned short* whh_lds = (unsigned short*)smem;                 // 128*1024*2 = 256KB
  float* gate_lds = (float*)(smem + 128 * 1024 * 2);               // [128][32]  =  16KB
  float* c_lds    = gate_lds + 128 * 32;                           // [32][32]   =   4KB

  const int tid  = threadIdx.x;
  const int lane = tid & 31;
  const int wave = tid >> 5;                  // 0..7 : local gate-row tile (16 rows)
  const int j0   = blockIdx.x * 32;

  // Async-DMA this WG's W_hh slice into LDS (local row lr = g*32+jj <- global
  // row g*1024 + j0 + jj). LDS byte address = low 32 bits of the generic
  // pointer (ISA 10.2: LDS aperture truncates to addr[31:0]); each lane issues
  // 16B transfers tracked by ASYNCcnt — no VGPR data movement.
  {
    const unsigned lds_base = (unsigned)(size_t)(void*)whh_lds;
    for (int idx = tid; idx < 128 * 1024 / 8; idx += 256) {
      int lr = idx >> 7, kk = (idx & 127) * 8;
      int g = lr >> 5, jj = lr & 31;
      const unsigned short* gp = whh + ((long)(g * H_ + j0 + jj)) * H_ + kk;
      unsigned loff = lds_base + (unsigned)idx * 16u;
      asm volatile("global_load_async_to_lds_b128 %0, %1, off"
                   :: "v"(loff), "v"(gp) : "memory");
    }
    asm volatile("s_wait_asynccnt 0x0" ::: "memory");
  }
  for (int e = tid; e < 32 * 32; e += 256) c_lds[e] = 0.f;
  __syncthreads();

  for (int t = 0; t < S_; ++t) {
    v8f acc0 = v8f{}, acc1 = v8f{};
    if (t > 0) {                               // h_{-1} == 0 -> gates = xg only at t==0
      const int bA = lane & 15;
      const long hr0 = hmode ? (long)(((t - 1) & 1) * 32 + bA)      : ((long)bA * S_ + (t - 1));
      const long hr1 = hmode ? (long)(((t - 1) & 1) * 32 + bA + 16) : ((long)(bA + 16) * S_ + (t - 1));
      const int kofA = (lane >> 4) ? 8 : 0;
      const unsigned short* lb = &whh_lds[(wave * 16 + (lane & 15)) * H_ + ((lane >> 4) ? 8 : 0)];
#pragma unroll 4
      for (int k = 0; k < H_; k += 32) {
        TileBf a0, a1, bt;
        const unsigned short* p0 = hsrc + hr0 * H_ + k + kofA;
        a0.q[0] = *(const uint4*)p0; a0.q[1] = *(const uint4*)(p0 + 16);
        const unsigned short* p1 = hsrc + hr1 * H_ + k + kofA;
        a1.q[0] = *(const uint4*)p1; a1.q[1] = *(const uint4*)(p1 + 16);
        const unsigned short* pb = lb + k;
        bt.q[0] = *(const uint4*)pb; bt.q[1] = *(const uint4*)(pb + 16);
        acc0 = __builtin_amdgcn_wmma_f32_16x16x32_bf16(false, a0.v, false, bt.v,
                                                       (short)0, acc0, false, false);
        acc1 = __builtin_amdgcn_wmma_f32_16x16x32_bf16(false, a1.v, false, bt.v,
                                                       (short)0, acc1, false, false);
      }
    }
    // spill gate accumulators: gate_lds[lr][b],  lr = local gate row, b = batch
    {
      const int lr = wave * 16 + (lane & 15);
      const int mh = (lane >> 4) * 8;
#pragma unroll
      for (int r = 0; r < 8; ++r) {
        gate_lds[lr * 32 + (r + mh)]      = acc0[r];
        gate_lds[lr * 32 + (16 + r + mh)] = acc1[r];
      }
    }
    __syncthreads();

    // pointwise LSTM: 32 b x 32 j = 1024 elems, 4 per thread (coalesced in jj)
#pragma unroll
    for (int e = 0; e < 4; ++e) {
      const int el = tid + 256 * e;
      const int b = el >> 5, jj = el & 31;
      const long xrow = ((long)b * S_ + t) * G4_ + (j0 + jj);
      float gi = gate_lds[(0 * 32 + jj) * 32 + b] + bf16_f32(xg[xrow]);
      float gf = gate_lds[(1 * 32 + jj) * 32 + b] + bf16_f32(xg[xrow + H_]);
      float gg = gate_lds[(2 * 32 + jj) * 32 + b] + bf16_f32(xg[xrow + 2 * H_]);
      float go = gate_lds[(3 * 32 + jj) * 32 + b] + bf16_f32(xg[xrow + 3 * H_]);
      if (t + 1 < S_) __builtin_prefetch(&xg[xrow + G4_], 0, 0);   // global_prefetch next step
      float si = 1.f / (1.f + __expf(-gi));
      float sf = 1.f / (1.f + __expf(-gf));
      float so = 1.f / (1.f + __expf(-go));
      float cn = sf * c_lds[el] + si * tanhf(gg);
      c_lds[el] = cn;
      float h = so * tanhf(cn);
      const long orow = ((long)b * S_ + t) * H_ + (j0 + jj);
      out_h[orow] = h;
      out_c[orow] = cn;
      if (out_h2) out_h2[orow] = h;
      const long hw = hmode ? (long)((t & 1) * 32 + b) : ((long)b * S_ + t);
      hdst[hw * H_ + (j0 + jj)] = f32_bf16(h);
    }

    // grid-wide generation barrier across the 32 persistent WGs
    __threadfence();
    __syncthreads();
    if (tid == 0) {
      atomicAdd(bar, 1u);
      const unsigned target = (unsigned)(RWG_ * (t + 1));
      while (__hip_atomic_load(bar, __ATOMIC_RELAXED, __HIP_MEMORY_SCOPE_AGENT) < target)
        __builtin_amdgcn_s_sleep(1);
    }
    __syncthreads();
    __threadfence();
  }
}

// ---------------- host launcher ----------------
static inline size_t alignup(size_t x) { return (x + 255) & ~(size_t)255; }

extern "C" void kernel_launch(void* const* d_in, const int* in_sizes, int n_in,
                              void* d_out, int out_size, void* d_ws, size_t ws_size,
                              hipStream_t stream) {
  const float* states = (const float*)d_in[0];
  const int*   act    = (const int*)d_in[1];
  const float* Win    = (const float*)d_in[2];
  const float* bin    = (const float*)d_in[3];
  const float* emb    = (const float*)d_in[4];
  const float* Wih    = (const float*)d_in[5];
  const float* Whh    = (const float*)d_in[6];
  const float* bih    = (const float*)d_in[7];
  const float* bhh    = (const float*)d_in[8];
  float* out = (float*)d_out;

  const long long BSH = (long long)MROWS_ * H_;   // 16,777,216
  float* out_rnn = out;
  float* out_hs  = out + BSH;                      // (L,B,S,H)
  float* out_cs  = out + 3 * BSH;

  // workspace carve-up (~253 MB)
  char* p = (char*)d_ws;  size_t off = 0;
  unsigned short* st_bf  = (unsigned short*)(p + off); off = alignup(off + (size_t)MROWS_ * IN_ * 2);
  unsigned short* win_bf = (unsigned short*)(p + off); off = alignup(off + (size_t)PROJ_ * IN_ * 2);
  unsigned short* wih_bf = (unsigned short*)(p + off); off = alignup(off + (size_t)2 * G4_ * INNER_ * 2);
  unsigned short* whh_bf = (unsigned short*)(p + off); off = alignup(off + (size_t)2 * G4_ * H_ * 2);
  unsigned short* x0     = (unsigned short*)(p + off); off = alignup(off + (size_t)MROWS_ * INNER_ * 2);
  unsigned short* x1     = (unsigned short*)(p + off); off = alignup(off + (size_t)MROWS_ * H_ * 2);
  unsigned short* xg     = (unsigned short*)(p + off); off = alignup(off + (size_t)MROWS_ * G4_ * 2);
  unsigned short* hping  = (unsigned short*)(p + off); off = alignup(off + (size_t)2 * B_ * H_ * 2);
  unsigned int*   bar    = (unsigned int*)(p + off);   off = alignup(off + 256);

  // 1) bf16 conversions of activations + weights
  k_cvt_bf16<<<1024, 256, 0, stream>>>(states, st_bf, (long)MROWS_ * IN_);
  k_cvt_bf16<<<256, 256, 0, stream>>>(Win, win_bf, (long)PROJ_ * IN_);
  k_cvt_bf16<<<1024, 256, 0, stream>>>(Wih, wih_bf, (long)2 * G4_ * INNER_);
  k_cvt_bf16<<<1024, 256, 0, stream>>>(Whh, whh_bf, (long)2 * G4_ * H_);
  // 2) embedding gather into x0[:, 960:1024]
  k_embed<<<512, 256, 0, stream>>>(act, emb, x0);
  // 3) input projection: x0[:, :960] = relu(states @ Win^T + bin)   (WMMA)
  {
    dim3 g(PROJ_ / 64, MROWS_ / 128);
    k_gemm_bf16<<<g, 128, 0, stream>>>(st_bf, IN_, win_bf, IN_, bin, nullptr,
                                       x0, INNER_, IN_, 1);
  }

  const size_t recur_smem = 128 * 1024 * 2 + 128 * 32 * 4 + 32 * 32 * 4;  // 276 KB
  for (int l = 0; l < 2; ++l) {
    const unsigned short* xin = (l == 0) ? x0 : x1;
    // 4) xg = x @ Wih[l]^T + (b_ih + b_hh)   (WMMA, M=16384 N=4096 K=1024)
    dim3 g(G4_ / 64, MROWS_ / 128);
    k_gemm_bf16<<<g, 128, 0, stream>>>(xin, INNER_,
                                       wih_bf + (size_t)l * G4_ * INNER_, INNER_,
                                       bih + (size_t)l * G4_, bhh + (size_t)l * G4_,
                                       xg, G4_, INNER_, 0);
    // 5) persistent recurrence over 512 steps
    k_zero_bar<<<1, 32, 0, stream>>>(bar);
    if (l == 0) {
      k_lstm_recur<<<RWG_, 256, recur_smem, stream>>>(
          xg, whh_bf, x1, x1, /*hmode=*/0,
          out_hs, out_cs, nullptr, bar);
    } else {
      k_lstm_recur<<<RWG_, 256, recur_smem, stream>>>(
          xg, whh_bf + (size_t)G4_ * H_, hping, hping, /*hmode=*/1,
          out_hs + BSH, out_cs + BSH, out_rnn, bar);
    }
  }
  (void)in_sizes; (void)n_in; (void)out_size; (void)ws_size;
}
